// BlockGRU_16054587752461
// MI455X (gfx1250) — compile-verified, matrix operand order checked by reading, not run on hardware
//
#include <hip/hip_runtime.h>
#include <hip/hip_bf16.h>

#define INPUT_DIM  1024
#define HIDDEN_DIM 2048
#define NUM_BLOCKS 8
#define BLOCK_SZ   256
#define GATES3     768
#define BATCH      2048

#define KCH        64                  // K elements staged per barrier interval
#define NROWS_B    192                 // 3 gates * 64 cols staged per chunk
#define BPAD       144                 // LDS bytes per staged row (128 data + 16 pad)
#define LDSBUF     (NROWS_B * BPAD)    // 27648 B per buffer, double buffered

typedef __attribute__((ext_vector_type(16))) __bf16        v16bf;
typedef __attribute__((ext_vector_type(8)))  float         v8f;
typedef __attribute__((ext_vector_type(4)))  unsigned int  u32x4;
typedef __attribute__((ext_vector_type(4)))  int           i32x4;

union FragU { v16bf v; u32x4 q[2]; };

#if __has_builtin(__builtin_amdgcn_global_load_async_to_lds_b128)
#define USE_ASYNC_COPY 1
typedef __attribute__((address_space(1))) i32x4 gvec_t;   // global int4
typedef __attribute__((address_space(3))) i32x4 lvec_t;   // LDS int4
#else
#define USE_ASYNC_COPY 0
#endif

// ---- 16B global -> LDS copy (async-to-LDS on gfx1250 when available) ----
__device__ __forceinline__ void copy16(const void* gsrc, void* ldst) {
#if USE_ASYNC_COPY
  void* g = const_cast<void*>(gsrc);
  __builtin_amdgcn_global_load_async_to_lds_b128((gvec_t*)g, (lvec_t*)ldst, 0, 0);
#else
  *(u32x4*)ldst = *(const u32x4*)gsrc;
#endif
}

__device__ __forceinline__ void async_fence_and_barrier() {
#if USE_ASYNC_COPY
#if __has_builtin(__builtin_amdgcn_s_wait_asynccnt)
  __builtin_amdgcn_s_wait_asynccnt(0);
#else
  asm volatile("s_wait_asynccnt 0x0" ::: "memory");
#endif
#endif
  __syncthreads();
}

// Load a 16-element bf16 K-strip WMMA operand from global (row-major, rowStride elems).
// 16x16x32 bf16 lane layout: lanes 0-15 hold K={k0..k0+7, k0+16..k0+23},
// lanes 16-31 hold K={k0+8..k0+15, k0+24..k0+31}.
__device__ __forceinline__ v16bf load_frag(const __bf16* __restrict__ base,
                                           int rowStride, int row, int k0, int khalf) {
  const __bf16* p = base + (long)row * rowStride + k0 + khalf * 8;
  FragU f;
  f.q[0] = *(const u32x4*)(p);
  f.q[1] = *(const u32x4*)(p + 16);
  return f.v;
}

// Same strip but from a staged LDS chunk: row = gate*64 + colLocal, 128B of K data + pad.
__device__ __forceinline__ v16bf lds_frag(const unsigned char* __restrict__ buf,
                                          int row, int kstep, int khalf) {
  const unsigned char* p = buf + row * BPAD + kstep * 64 + khalf * 16;
  FragU f;
  f.q[0] = *(const u32x4*)(p);
  f.q[1] = *(const u32x4*)(p + 32);
  return f.v;
}

__device__ __forceinline__ v8f wmma_bf16(v16bf a, v16bf b, v8f c) {
  return __builtin_amdgcn_wmma_f32_16x16x32_bf16(false, a, false, b, (short)0, c, false, false);
}

__device__ __forceinline__ float fast_sigmoid(float x) {
  return 1.0f / (1.0f + __expf(-x));
}
__device__ __forceinline__ float fast_tanh(float x) {
  float t = __expf(-2.0f * x);
  return (1.0f - t) / (1.0f + t);
}

// Stage one K-chunk of B: 192 rows (3 gates x 64 cols) x 128B, cooperatively by 256 threads.
template <int KDIM>
__device__ __forceinline__ void stage_B(const __bf16* __restrict__ Wn, int cgBase, int kc0,
                                        unsigned char* __restrict__ buf, int tid) {
#pragma unroll
  for (int i = 0; i < 6; ++i) {
    int p    = tid + i * 256;        // 0..1535 16B pieces
    int r    = p >> 3;               // staged row 0..191
    int sub  = p & 7;                // 16B piece within 128B row
    int gate = r >> 6;               // 0..2
    int colL = r & 63;               // column within col-group
    const __bf16* src = Wn + (size_t)(gate * BLOCK_SZ + cgBase + colL) * KDIM + kc0 + sub * 8;
    copy16(src, buf + r * BPAD + sub * 16);
  }
}

// One staged K-chunk of GEMM: 2 k-steps x 4 col-tiles x 3 gates = 24 WMMAs.
__device__ __forceinline__ void gemm_chunk(const __bf16* __restrict__ A, int astride, int rowA,
                                           int aK0, const unsigned char* __restrict__ buf,
                                           int nIdx, int khalf,
                                           v8f (&aR)[4], v8f (&aZ)[4], v8f (&aN)[4]) {
#pragma unroll
  for (int ks = 0; ks < 2; ++ks) {
    v16bf a = load_frag(A, astride, rowA, aK0 + ks * 32, khalf);
#pragma unroll
    for (int t = 0; t < 4; ++t) {
      const int rowB = t * 16 + nIdx;
      v16bf b0 = lds_frag(buf, rowB,       ks, khalf);
      v16bf b1 = lds_frag(buf, 64 + rowB,  ks, khalf);
      v16bf b2 = lds_frag(buf, 128 + rowB, ks, khalf);
      aR[t] = wmma_bf16(a, b0, aR[t]);
      aZ[t] = wmma_bf16(a, b1, aZ[t]);
      aN[t] = wmma_bf16(a, b2, aN[t]);
    }
  }
}

// ---------------------------------------------------------------- conversion
__global__ void cvt_f32_to_bf16(const float* __restrict__ src,
                                __hip_bfloat16* __restrict__ dst, int n4) {
  int i = blockIdx.x * blockDim.x + threadIdx.x;
  if (i >= n4) return;
  float4 f = ((const float4*)src)[i];
  union { __hip_bfloat16 h[4]; uint2 u; } o;
  o.h[0] = __float2bfloat16(f.x);
  o.h[1] = __float2bfloat16(f.y);
  o.h[2] = __float2bfloat16(f.z);
  o.h[3] = __float2bfloat16(f.w);
  ((uint2*)dst)[i] = o.u;
}

// ---------------------------------------------------------------- main kernel
// grid = (BATCH/128, NUM_BLOCKS), block = 256 threads (8 waves, wave32).
// Wave w owns 16 rows; 4 col-groups of 64 cols; B staged via (async) LDS double buffer.
__global__ __launch_bounds__(256)
void block_gru_wmma(const __bf16* __restrict__ inBF,   // [2048,1024]
                    const __bf16* __restrict__ hBF,    // [2048,2048]
                    const __bf16* __restrict__ WihBF,  // [8,768,1024]
                    const __bf16* __restrict__ WhhBF,  // [8,768,256]
                    const float*  __restrict__ bih,    // [8,768]
                    const float*  __restrict__ bhh,    // [8,768]
                    const float*  __restrict__ hOld,   // [2048,2048] f32
                    float*        __restrict__ out)    // [2048,2048] f32
{
  __shared__ __align__(16) unsigned char ldsB[2][LDSBUF];

  const int tid   = threadIdx.x;
  const int lane  = tid & 31;
  const int wave  = tid >> 5;
  const int bt    = blockIdx.x;
  const int nb    = blockIdx.y;
  const int nIdx  = lane & 15;
  const int khalf = lane >> 4;
  const int rowA  = bt * 128 + wave * 16 + nIdx;

  const __bf16* Wih_n = WihBF + (long)nb * GATES3 * INPUT_DIM;
  const __bf16* Whh_n = WhhBF + (long)nb * GATES3 * BLOCK_SZ;
  const float*  bih_n = bih + nb * GATES3;
  const float*  bhh_n = bhh + nb * GATES3;

  for (int cg = 0; cg < BLOCK_SZ / 64; ++cg) {
    const int cgBase = cg * 64;

    v8f accR[4], accZ[4], accIN[4], accHN[4];
#pragma unroll
    for (int t = 0; t < 4; ++t) {
      const int col = cgBase + t * 16 + nIdx;
      const float bR0 = bih_n[col]       + bhh_n[col];
      const float bZ0 = bih_n[256 + col] + bhh_n[256 + col];
      const float bI0 = bih_n[512 + col];
      const float bH0 = bhh_n[512 + col];
#pragma unroll
      for (int j = 0; j < 8; ++j) {
        accR[t][j] = bR0; accZ[t][j] = bZ0; accIN[t][j] = bI0; accHN[t][j] = bH0;
      }
    }

    // ---- GEMM 1: gi = x @ W_ih[nb]^T  (K = 1024, 16 staged chunks) ----
    stage_B<INPUT_DIM>(Wih_n, cgBase, 0, ldsB[0], tid);
    async_fence_and_barrier();
    for (int kc = 0; kc < INPUT_DIM / KCH; ++kc) {
      const unsigned char* cur = ldsB[kc & 1];
      if (kc + 1 < INPUT_DIM / KCH)
        stage_B<INPUT_DIM>(Wih_n, cgBase, (kc + 1) * KCH, ldsB[(kc + 1) & 1], tid);
      gemm_chunk(inBF, INPUT_DIM, rowA, kc * KCH, cur, nIdx, khalf, accR, accZ, accIN);
      async_fence_and_barrier();
    }

    // ---- GEMM 2: gh = h_nb @ W_hh[nb]^T  (K = 256, 4 staged chunks) ----
    stage_B<BLOCK_SZ>(Whh_n, cgBase, 0, ldsB[0], tid);
    async_fence_and_barrier();
    for (int kc = 0; kc < BLOCK_SZ / KCH; ++kc) {
      const unsigned char* cur = ldsB[kc & 1];
      if (kc + 1 < BLOCK_SZ / KCH)
        stage_B<BLOCK_SZ>(Whh_n, cgBase, (kc + 1) * KCH, ldsB[(kc + 1) & 1], tid);
      gemm_chunk(hBF, HIDDEN_DIM, rowA, nb * BLOCK_SZ + kc * KCH, cur, nIdx, khalf,
                 accR, accZ, accHN);
      async_fence_and_barrier();
    }

    // ---- fused GRU pointwise + store (C/D layout: VGPR j -> M=j+8*khalf, N=lane&15) ----
#pragma unroll
    for (int t = 0; t < 4; ++t) {
      const int colG = nb * BLOCK_SZ + cgBase + t * 16 + nIdx;
#pragma unroll
      for (int j = 0; j < 8; ++j) {
        const int row = bt * 128 + wave * 16 + j + 8 * khalf;
        float r  = fast_sigmoid(accR[t][j]);
        float z  = fast_sigmoid(accZ[t][j]);
        float nn = fast_tanh(accIN[t][j] + r * accHN[t][j]);
        float h  = hOld[(long)row * HIDDEN_DIM + colG];
        out[(long)row * HIDDEN_DIM + colG] = (1.0f - z) * nn + z * h;
      }
    }
  }
}

// ---------------------------------------------------------------- host launch
extern "C" void kernel_launch(void* const* d_in, const int* in_sizes, int n_in,
                              void* d_out, int out_size, void* d_ws, size_t ws_size,
                              hipStream_t stream) {
  const float* input  = (const float*)d_in[0];
  const float* hidden = (const float*)d_in[1];
  const float* W_ih   = (const float*)d_in[2];
  const float* W_hh   = (const float*)d_in[3];
  const float* b_ih   = (const float*)d_in[4];
  const float* b_hh   = (const float*)d_in[5];
  float* out = (float*)d_out;

  // bf16 staging in workspace (~27.8 MB), all offsets 16B aligned
  __hip_bfloat16* inBF  = (__hip_bfloat16*)d_ws;
  __hip_bfloat16* hBF   = inBF  + (size_t)BATCH * INPUT_DIM;
  __hip_bfloat16* WihBF = hBF   + (size_t)BATCH * HIDDEN_DIM;
  __hip_bfloat16* WhhBF = WihBF + (size_t)NUM_BLOCKS * GATES3 * INPUT_DIM;

  auto cvt = [&](const float* s, __hip_bfloat16* d, size_t n) {
    int n4 = (int)(n / 4);
    cvt_f32_to_bf16<<<(n4 + 255) / 256, 256, 0, stream>>>(s, d, n4);
  };
  cvt(input,  inBF,  (size_t)BATCH * INPUT_DIM);
  cvt(hidden, hBF,   (size_t)BATCH * HIDDEN_DIM);
  cvt(W_ih,   WihBF, (size_t)NUM_BLOCKS * GATES3 * INPUT_DIM);
  cvt(W_hh,   WhhBF, (size_t)NUM_BLOCKS * GATES3 * BLOCK_SZ);

  dim3 grid(BATCH / 128, NUM_BLOCKS);
  block_gru_wmma<<<grid, 256, 0, stream>>>(
      (const __bf16*)inBF, (const __bf16*)hBF,
      (const __bf16*)WihBF, (const __bf16*)WhhBF,
      b_ih, b_hh, hidden, out);
}
